// SlidingWindowAttention_50130858279277
// MI455X (gfx1250) — matrix operand order repaired and checked by reading, not compiled, not run on hardware
//
#include <hip/hip_runtime.h>
#include <math.h>

typedef __bf16 bf16_t;
typedef __bf16 v4bf  __attribute__((ext_vector_type(4)));
typedef __bf16 v8bf  __attribute__((ext_vector_type(8)));
typedef __bf16 v16bf __attribute__((ext_vector_type(16)));
typedef float  v8f   __attribute__((ext_vector_type(8)));

#define HIDDEN 1024
#define NHEADS 16
#define HDIM   64
#define SEQ    2048
#define BATCH  2
#define MROWS  (BATCH * SEQ)  // 4096

// ---- WMMA helpers ---------------------------------------------------------

__device__ __forceinline__ v8f wmma_bf16(v16bf a, v16bf b, v8f c) {
  // D = A(16x32 bf16) x B(32x16 bf16) + C(16x16 f32)
  return __builtin_amdgcn_wmma_f32_16x16x32_bf16(
      /*neg_a=*/false, a, /*neg_b=*/false, b,
      /*c_mod=*/(short)0, c, /*reuse_a=*/false, /*reuse_b=*/false);
}

// A-operand fragment: lane<16 holds K {0..7,16..23}; lane>=16 holds K {8..15,24..31}.
__device__ __forceinline__ v16bf load_a(const bf16_t* p, int kb) {
  v8bf lo = *(const v8bf*)(p + kb);
  v8bf hi = *(const v8bf*)(p + kb + 16);
  return __builtin_shufflevector(lo, hi, 0, 1, 2, 3, 4, 5, 6, 7,
                                 8, 9, 10, 11, 12, 13, 14, 15);
}

// B-operand fragment: lane<16 holds K {0..7,16..23}; lane>=16 holds K {8..15,24..31}.
__device__ __forceinline__ v16bf load_b(const bf16_t* p, int kbB) {
  v8bf lo = *(const v8bf*)(p + (kbB >> 1));
  v8bf hi = *(const v8bf*)(p + (kbB >> 1) + 16);
  return __builtin_shufflevector(lo, hi, 0, 1, 2, 3, 4, 5, 6, 7,
                                 8, 9, 10, 11, 12, 13, 14, 15);
}

// ---- GEMM: Y = X(f32)[M,K] * W(f32)[N,K]^T, output bf16 head layout -------
// Block tile 128x128, 4 waves, each wave a 64x64 quadrant (16 WMMA / K-step).

__global__ __launch_bounds__(128)
void gemm_qkv_kernel(const float* __restrict__ X, const float* __restrict__ W,
                     bf16_t* __restrict__ Out) {
  __shared__ __attribute__((aligned(16))) bf16_t As[128 * 40];
  __shared__ __attribute__((aligned(16))) bf16_t Bs[128 * 40];

  const int tid  = threadIdx.x;
  const int lane = tid & 31, wid = tid >> 5;
  const int lr   = lane & 15;
  const int kbA  = (lane < 16) ? 0 : 8;
  const int kbB  = (lane < 16) ? 0 : 16;
  const int hal8 = (lane >> 4) << 3;
  const int mw = (wid >> 1) * 64;   // wave quadrant in M
  const int nw = (wid & 1) * 64;    // wave quadrant in N
  const int m0 = blockIdx.y * 128;
  const int n0 = blockIdx.x * 128;

  v8f acc[4][4] = {};

  for (int kt = 0; kt < HIDDEN / 32; ++kt) {
    // Stage A,B tiles (128x32 f32 -> bf16), float4 global loads, b64 LDS stores.
    const float4* Xg = (const float4*)(X + (size_t)m0 * HIDDEN + kt * 32);
    const float4* Wg = (const float4*)(W + (size_t)n0 * HIDDEN + kt * 32);
#pragma unroll
    for (int i = 0; i < 8; ++i) {
      int idx = tid + i * 128;           // 0..1023
      int m = idx >> 3, k4 = idx & 7;    // row, float4 index within 32-col slab
      float4 fa = Xg[(size_t)m * (HIDDEN / 4) + k4];
      float4 fb = Wg[(size_t)m * (HIDDEN / 4) + k4];
      *(v4bf*)&As[m * 40 + k4 * 4] =
          (v4bf){(bf16_t)fa.x, (bf16_t)fa.y, (bf16_t)fa.z, (bf16_t)fa.w};
      *(v4bf*)&Bs[m * 40 + k4 * 4] =
          (v4bf){(bf16_t)fb.x, (bf16_t)fb.y, (bf16_t)fb.z, (bf16_t)fb.w};
    }
    // Prefetch next K-slab while this one is consumed (global_prefetch_b8).
    if (kt + 1 < HIDDEN / 32) {
      __builtin_prefetch(X + (size_t)(m0 + tid) * HIDDEN + (kt + 1) * 32, 0, 1);
      __builtin_prefetch(W + (size_t)(n0 + tid) * HIDDEN + (kt + 1) * 32, 0, 1);
    }
    __syncthreads();

    v16bf af[4], bfv[4];
#pragma unroll
    for (int i = 0; i < 4; ++i)
      af[i] = load_a(&As[(mw + i * 16 + lr) * 40], kbA);
#pragma unroll
    for (int j = 0; j < 4; ++j)
      bfv[j] = load_b(&Bs[(nw + j * 16 + lr) * 40], kbB);
#pragma unroll
    for (int i = 0; i < 4; ++i)
#pragma unroll
      for (int j = 0; j < 4; ++j)
        acc[i][j] = wmma_bf16(af[i], bfv[j], acc[i][j]);
    __syncthreads();
  }

  // Store bf16 into [B*nH][S][HDIM] layout for attention.
#pragma unroll
  for (int i = 0; i < 4; ++i)
#pragma unroll
    for (int j = 0; j < 4; ++j)
#pragma unroll
      for (int r = 0; r < 8; ++r) {
        int m = m0 + mw + i * 16 + r + hal8;
        int n = n0 + nw + j * 16 + lr;
        int b = m >> 11, s = m & 2047, h = n >> 6, d = n & 63;
        Out[((size_t)(b * NHEADS + h) * SEQ + s) * HDIM + d] = (bf16_t)acc[i][j][r];
      }
}

// ---- GEMM: Out(f32)[M,N] = X(bf16)[M,K] * W(f32)[N,K]^T -------------------

__global__ __launch_bounds__(128)
void gemm_out_kernel(const bf16_t* __restrict__ X, const float* __restrict__ W,
                     float* __restrict__ Out) {
  __shared__ __attribute__((aligned(16))) bf16_t As[128 * 40];
  __shared__ __attribute__((aligned(16))) bf16_t Bs[128 * 40];

  const int tid  = threadIdx.x;
  const int lane = tid & 31, wid = tid >> 5;
  const int lr   = lane & 15;
  const int kbA  = (lane < 16) ? 0 : 8;
  const int kbB  = (lane < 16) ? 0 : 16;
  const int hal8 = (lane >> 4) << 3;
  const int mw = (wid >> 1) * 64;
  const int nw = (wid & 1) * 64;
  const int m0 = blockIdx.y * 128;
  const int n0 = blockIdx.x * 128;

  v8f acc[4][4] = {};

  for (int kt = 0; kt < HIDDEN / 32; ++kt) {
    // A is already bf16: copy 128x32 tile with uint4 (8 bf16) transfers.
    const uint4* Xg = (const uint4*)(X + (size_t)m0 * HIDDEN + kt * 32);
#pragma unroll
    for (int i = 0; i < 4; ++i) {
      int idx = tid + i * 128;          // 0..511
      int m = idx >> 2, q = idx & 3;    // row, uint4 index (8 bf16 each)
      uint4 v = Xg[(size_t)m * (HIDDEN * 2 / 16) + q];
      *(uint4*)&As[m * 40 + q * 8] = v;
    }
    const float4* Wg = (const float4*)(W + (size_t)n0 * HIDDEN + kt * 32);
#pragma unroll
    for (int i = 0; i < 8; ++i) {
      int idx = tid + i * 128;
      int m = idx >> 3, k4 = idx & 7;
      float4 fb = Wg[(size_t)m * (HIDDEN / 4) + k4];
      *(v4bf*)&Bs[m * 40 + k4 * 4] =
          (v4bf){(bf16_t)fb.x, (bf16_t)fb.y, (bf16_t)fb.z, (bf16_t)fb.w};
    }
    if (kt + 1 < HIDDEN / 32) {
      __builtin_prefetch(X + (size_t)(m0 + tid) * HIDDEN + (kt + 1) * 32, 0, 1);
      __builtin_prefetch(W + (size_t)(n0 + tid) * HIDDEN + (kt + 1) * 32, 0, 1);
    }
    __syncthreads();

    v16bf af[4], bfv[4];
#pragma unroll
    for (int i = 0; i < 4; ++i)
      af[i] = load_a(&As[(mw + i * 16 + lr) * 40], kbA);
#pragma unroll
    for (int j = 0; j < 4; ++j)
      bfv[j] = load_b(&Bs[(nw + j * 16 + lr) * 40], kbB);
#pragma unroll
    for (int i = 0; i < 4; ++i)
#pragma unroll
      for (int j = 0; j < 4; ++j)
        acc[i][j] = wmma_bf16(af[i], bfv[j], acc[i][j]);
    __syncthreads();
  }

#pragma unroll
  for (int i = 0; i < 4; ++i)
#pragma unroll
    for (int j = 0; j < 4; ++j)
#pragma unroll
      for (int r = 0; r < 8; ++r) {
        int m = m0 + mw + i * 16 + r + hal8;
        int n = n0 + nw + j * 16 + lr;
        Out[(size_t)m * HIDDEN + n] = acc[i][j][r];
      }
}

// ---- Flash attention: per block = 64 query rows of one head ---------------

__global__ __launch_bounds__(128)
void attn_kernel(const bf16_t* __restrict__ Q, const bf16_t* __restrict__ K,
                 const bf16_t* __restrict__ V, bf16_t* __restrict__ O) {
  __shared__ __attribute__((aligned(16))) bf16_t Ks[64 * 72];   // K rows (Bt for QK^T)
  __shared__ __attribute__((aligned(16))) bf16_t Vts[64 * 72];  // V^T rows (Bt for PV)
  __shared__ __attribute__((aligned(16))) float  Ss[4][16 * 68];
  __shared__ __attribute__((aligned(16))) bf16_t Ps[4][16 * 72];
  __shared__ float m_s[4][16];
  __shared__ float l_s[4][16];
  __shared__ float f_s[4][16];

  const int tid  = threadIdx.x;
  const int lane = tid & 31, wid = tid >> 5;
  const int lr   = lane & 15;
  const int kbA  = (lane < 16) ? 0 : 8;
  const int kbB  = (lane < 16) ? 0 : 16;
  const int hal8 = (lane >> 4) << 3;
  const int bh   = blockIdx.y;       // batch*head
  const int q0   = blockIdx.x * 64;  // query base row
  const int qrow = q0 + wid * 16 + lr;

  // Q fragments for d chunks [0..31], [32..63]
  const bf16_t* qp = Q + ((size_t)bh * SEQ + qrow) * HDIM;
  v16bf aq0 = load_a(qp + 0,  kbA);
  v16bf aq1 = load_a(qp + 32, kbA);

  v8f o[4] = {};
  if (lane < 16) { m_s[wid][lr] = -INFINITY; l_s[wid][lr] = 0.0f; }
  __syncthreads();

  const bf16_t* Kg0 = K + (size_t)bh * SEQ * HDIM;
  const bf16_t* Vg0 = V + (size_t)bh * SEQ * HDIM;

  for (int jb = 0; jb < SEQ / 64; ++jb) {
    // Cooperative loads: K tile row-major, V tile transposed.
    const unsigned int* Kg = (const unsigned int*)(Kg0 + (size_t)jb * 64 * HDIM);
    for (int idx = tid; idx < 64 * 32; idx += 128) {
      int n = idx >> 5, dw = idx & 31;
      *(unsigned int*)&Ks[n * 72 + dw * 2] = Kg[n * 32 + dw];
    }
    const bf16_t* Vg = Vg0 + (size_t)jb * 64 * HDIM;
    for (int idx = tid; idx < 64 * 64; idx += 128) {
      int n = idx >> 6, d = idx & 63;
      Vts[d * 72 + n] = Vg[n * 64 + d];
    }
    __syncthreads();

    // Scores: S = (Q K^T) * 1/sqrt(64)
    for (int nt = 0; nt < 4; ++nt) {
      v16bf b0 = load_b(&Ks[(nt * 16 + lr) * 72 + 0],  kbB);
      v16bf b1 = load_b(&Ks[(nt * 16 + lr) * 72 + 32], kbB);
      v8f s = {};
      s = wmma_bf16(aq0, b0, s);
      s = wmma_bf16(aq1, b1, s);
      for (int r = 0; r < 8; ++r)
        Ss[wid][(r + hal8) * 68 + nt * 16 + lr] = s[r] * 0.125f;
    }
    __syncthreads();

    // Online softmax: lane lr (<16) owns query row lr of this wave's tile.
    if (lane < 16) {
      float mo = m_s[wid][lr];
      const float* srow = &Ss[wid][lr * 68];
      float bmax = -INFINITY;
      for (int c = 0; c < 64; ++c) bmax = fmaxf(bmax, srow[c]);
      float mn  = fmaxf(mo, bmax);
      float fac = __expf(mo - mn);
      float sum = 0.0f;
      for (int c = 0; c < 64; ++c) {
        float p = __expf(srow[c] - mn);
        Ps[wid][lr * 72 + c] = (bf16_t)p;
        sum += p;
      }
      m_s[wid][lr] = mn;
      l_s[wid][lr] = l_s[wid][lr] * fac + sum;
      f_s[wid][lr] = fac;
    }
    __syncthreads();

    // Rescale accumulators, then O += P * V
    {
      float fr[8];
      for (int r = 0; r < 8; ++r) fr[r] = f_s[wid][r + hal8];
      for (int dt = 0; dt < 4; ++dt)
        for (int r = 0; r < 8; ++r) o[dt][r] *= fr[r];

      v16bf p0 = load_a(&Ps[wid][lr * 72 + 0],  kbA);
      v16bf p1 = load_a(&Ps[wid][lr * 72 + 32], kbA);
      for (int dt = 0; dt < 4; ++dt) {
        v16bf v0 = load_b(&Vts[(dt * 16 + lr) * 72 + 0],  kbB);
        v16bf v1 = load_b(&Vts[(dt * 16 + lr) * 72 + 32], kbB);
        o[dt] = wmma_bf16(p0, v0, o[dt]);
        o[dt] = wmma_bf16(p1, v1, o[dt]);
      }
    }
    __syncthreads();
  }

  // Finalize: divide by row sums, store bf16 into [B*S][HIDDEN] for out-proj.
  float inv[8];
  for (int r = 0; r < 8; ++r) inv[r] = 1.0f / l_s[wid][r + hal8];
  const int b = bh >> 4, h = bh & 15;
  for (int dt = 0; dt < 4; ++dt)
    for (int r = 0; r < 8; ++r) {
      int m   = q0 + wid * 16 + r + hal8;
      int col = h * HDIM + dt * 16 + lr;
      O[(size_t)(b * SEQ + m) * HIDDEN + col] = (bf16_t)(o[dt][r] * inv[r]);
    }
}

// ---- Host launcher --------------------------------------------------------

extern "C" void kernel_launch(void* const* d_in, const int* in_sizes, int n_in,
                              void* d_out, int out_size, void* d_ws, size_t ws_size,
                              hipStream_t stream) {
  (void)in_sizes; (void)n_in; (void)out_size; (void)ws_size;

  const float* X  = (const float*)d_in[0];
  const float* Wq = (const float*)d_in[1];
  const float* Wk = (const float*)d_in[2];
  const float* Wv = (const float*)d_in[3];
  const float* Wo = (const float*)d_in[4];
  float* out = (float*)d_out;

  char* ws = (char*)d_ws;
  const size_t headBytes = (size_t)BATCH * NHEADS * SEQ * HDIM * sizeof(bf16_t);  // 8 MB
  bf16_t* Qb = (bf16_t*)(ws + 0 * headBytes);
  bf16_t* Kb = (bf16_t*)(ws + 1 * headBytes);
  bf16_t* Vb = (bf16_t*)(ws + 2 * headBytes);
  bf16_t* X2 = (bf16_t*)(ws + 3 * headBytes);

  dim3 gg(HIDDEN / 128, MROWS / 128);  // (8, 32)
  gemm_qkv_kernel<<<gg, 128, 0, stream>>>(X, Wq, Qb);
  gemm_qkv_kernel<<<gg, 128, 0, stream>>>(X, Wk, Kb);
  gemm_qkv_kernel<<<gg, 128, 0, stream>>>(X, Wv, Vb);
  attn_kernel<<<dim3(SEQ / 64, BATCH * NHEADS), 128, 0, stream>>>(Qb, Kb, Vb, X2);
  gemm_out_kernel<<<gg, 128, 0, stream>>>(X2, Wo, out);
}
